// MultiHeadedAttention_3152505996061
// MI455X (gfx1250) — compile-verified, matrix operand order checked by reading, not run on hardware
//
#include <hip/hip_runtime.h>
#include <hip/hip_bf16.h>

typedef __attribute__((ext_vector_type(16))) _Float16 v16h;
typedef __attribute__((ext_vector_type(8)))  _Float16 v8h;
typedef __attribute__((ext_vector_type(4)))  _Float16 v4h;
typedef __attribute__((ext_vector_type(8)))  float    v8f;

#define D_DIM  1024
#define S_DIM  2048
#define H_DIM  16
#define DK_DIM 64
#define NTOK   8192   // B*S = 4*2048

// ---------------------------------------------------------------------------
// Load a 16-bit WMMA A/B fragment from an LDS/global row.
// Per CDNA5 ISA 16-bit A-matrix 16x32 layout: lane holds row m=lane&15,
// kb=(lane>>4)*8, elements 0..7 = [kb..kb+7], elements 8..15 = [kb+16..kb+23].
// Caller passes p = &row[kb]; we read p[0..7] and p[16..23] (two b128 loads).
// ---------------------------------------------------------------------------
__device__ __forceinline__ v16h frag16(const _Float16* p) {
  v8h lo = *(const v8h*)(p);
  v8h hi = *(const v8h*)(p + 16);
  v16h r;
#pragma unroll
  for (int i = 0; i < 8; ++i) { r[i] = lo[i]; r[i + 8] = hi[i]; }
  return r;
}

__device__ __forceinline__ v8f wmma_f16(v16h a, v16h b, v8f c) {
  return __builtin_amdgcn_wmma_f32_16x16x32_f16(false, a, false, b, (short)0, c,
                                                false, false);
}

// Async global -> LDS copy, 16 bytes per lane (CDNA5 GLOBAL_LOAD_ASYNC_TO_LDS).
// Tracked by ASYNCcnt; caller must s_wait_asynccnt before consuming the LDS.
// Generic LDS pointers carry the LDS byte offset in the low 32 bits.
__device__ __forceinline__ void async_g2l_b128(void* lds, const void* gsrc) {
  const unsigned lds_off = (unsigned)(uintptr_t)lds;
  asm volatile("global_load_async_to_lds_b128 %0, %1, off"
               :: "v"(lds_off), "v"(gsrc) : "memory");
}
__device__ __forceinline__ void wait_asynccnt0() {
  asm volatile("s_wait_asynccnt 0x0" ::: "memory");
}

// 16-lane xor butterfly step. Prefer v_permlane16_b32 (pure VALU, no DS
// round-trip / dscnt waits); fall back to ds_bpermute-based __shfl_xor.
template <int M>
__device__ __forceinline__ float shfl_xor16(float x) {
#if __has_builtin(__builtin_amdgcn_permlane16)
  constexpr unsigned SEL0[9] = {0, 0x67452301u, 0x45670123u, 0, 0x32107654u,
                                0, 0, 0, 0xFEDCBA98u};
  constexpr unsigned SEL1[9] = {0, 0xEFCDAB89u, 0xCDEF89ABu, 0, 0xBA98FEDCu,
                                0, 0, 0, 0x76543210u};
  unsigned u = __float_as_uint(x);
  u = __builtin_amdgcn_permlane16(u, u, SEL0[M], SEL1[M], false, false);
  return __uint_as_float(u);
#else
  return __shfl_xor(x, M, 32);
#endif
}

// ---------------------------------------------------------------------------
// Tiled WMMA GEMM:  C[M,N] = A[M,K](f32) * (Bw .* Bmul)[K,N](f32) + bias
// 128x128 workgroup tile, BLK_K=32, 256 threads = 8 waves in 4(m) x 2(n).
// LDS double-buffered + register-staged prefetch: one barrier per k-step,
// global fetch of tile i+1 overlaps WMMA on tile i. B staged transposed ->
// Bs[n][k] so both fragment reads are contiguous ds_load_b128 pairs.
// ---------------------------------------------------------------------------
template <int OUT_F16, int HAS_MUL>
__global__ __launch_bounds__(256) void gemm_wmma(
    const float* __restrict__ A, const float* __restrict__ Bw,
    const float* __restrict__ Bmul, const float* __restrict__ bias,
    void* __restrict__ Cout, int M, int N, int K) {
  constexpr int LDA = 40;  // halves; 80B rows keep 16B alignment
  __shared__ __align__(16) _Float16 As[2][128 * LDA];
  __shared__ __align__(16) _Float16 Bs[2][128 * LDA];

  const int tid  = threadIdx.x;
  const int lane = tid & 31;
  const int wid  = tid >> 5;
  const int wm   = (wid & 3) * 32;   // wave m-offset in tile
  const int wn   = (wid >> 2) * 64;  // wave n-offset in tile
  const int m0   = blockIdx.y * 128;
  const int n0   = blockIdx.x * 128;

  v8f acc[2][4];
#pragma unroll
  for (int i = 0; i < 2; ++i)
#pragma unroll
    for (int j = 0; j < 4; ++j)
#pragma unroll
      for (int v = 0; v < 8; ++v) acc[i][j][v] = 0.0f;

  // staging coordinates
  const int ar  = tid >> 3;         // A: row within 32-row pass
  const int ac  = (tid & 7) * 4;    // A: k-chunk of 4
  const int bn1 = tid & 127;        // B: n column (lanes contiguous in n)
  const int bko = (tid >> 7) * 4;   // B: k sub-offset {0,4}

  const int fm = lane & 15;
  const int kb = (lane >> 4) * 8;

  float4 ra[4];
  float  rb[4][4];
  float  rbm[4][4];

  auto load_tile = [&](int k0) {
#pragma unroll
    for (int p = 0; p < 4; ++p)
      ra[p] = *(const float4*)(A + (size_t)(m0 + p * 32 + ar) * K + k0 + ac);
#pragma unroll
    for (int p = 0; p < 4; ++p) {
      const int kk = p * 8 + bko;
#pragma unroll
      for (int i = 0; i < 4; ++i) {
        const size_t off = (size_t)(k0 + kk + i) * N + n0 + bn1;
        rb[p][i] = Bw[off];
        if constexpr (HAS_MUL) rbm[p][i] = Bmul[off];
      }
    }
  };
  auto store_tile = [&](int buf) {
#pragma unroll
    for (int p = 0; p < 4; ++p) {
      v4h h;
      h[0] = (_Float16)ra[p].x; h[1] = (_Float16)ra[p].y;
      h[2] = (_Float16)ra[p].z; h[3] = (_Float16)ra[p].w;
      *(v4h*)(&As[buf][(p * 32 + ar) * LDA + ac]) = h;
    }
#pragma unroll
    for (int p = 0; p < 4; ++p) {
      const int kk = p * 8 + bko;
      v4h h;
#pragma unroll
      for (int i = 0; i < 4; ++i) {
        float x = rb[p][i];
        if constexpr (HAS_MUL) x *= rbm[p][i];
        h[i] = (_Float16)x;
      }
      *(v4h*)(&Bs[buf][bn1 * LDA + kk]) = h;
    }
  };

  load_tile(0);
  store_tile(0);

  int cur = 0;
  for (int k0 = 0; k0 < K; k0 += 32) {
    const bool more = (k0 + 32) < K;
    __syncthreads();               // buf[cur] visible; buf[cur^1] free
    if (more) load_tile(k0 + 32);  // global fetch overlaps WMMA below

    v16h a[2], b[4];
#pragma unroll
    for (int i = 0; i < 2; ++i)
      a[i] = frag16(&As[cur][(wm + i * 16 + fm) * LDA + kb]);
#pragma unroll
    for (int j = 0; j < 4; ++j)
      b[j] = frag16(&Bs[cur][(wn + j * 16 + fm) * LDA + kb]);
#pragma unroll
    for (int i = 0; i < 2; ++i)
#pragma unroll
      for (int j = 0; j < 4; ++j) acc[i][j] = wmma_f16(a[i], b[j], acc[i][j]);

    if (more) store_tile(cur ^ 1);
    cur ^= 1;
  }

  // ---- epilogue: C layout: lane holds col n=lane&15, rows m=v+8*(lane>>4)
  const int hf = lane >> 4;
#pragma unroll
  for (int j = 0; j < 4; ++j) {
    const int gn = n0 + wn + j * 16 + fm;
    const float bb = bias ? bias[gn] : 0.0f;
#pragma unroll
    for (int i = 0; i < 2; ++i) {
#pragma unroll
      for (int v = 0; v < 8; ++v) {
        const int gm = m0 + wm + i * 16 + hf * 8 + v;
        const float val = acc[i][j][v] + bb;
        if constexpr (OUT_F16)
          ((_Float16*)Cout)[(size_t)gm * N + gn] = (_Float16)val;
        else
          ((float*)Cout)[(size_t)gm * N + gn] = val;
      }
    }
  }
}

// ---------------------------------------------------------------------------
// Flash attention: one workgroup per (b, h, 128-query tile).
// 8 waves x 16 query rows. Double-buffered 64-key K/V tiles: tile kt+1 is
// staged (async K copy + V transpose) while tile kt is consumed, one barrier
// per iteration. QK^T and P*V via wmma_f32_16x16x32_f16; online softmax with
// v_permlane16 butterflies; 1/sqrt(DK) folded into the Q fragments.
// ---------------------------------------------------------------------------
__global__ __launch_bounds__(256) void attn_wmma(
    const _Float16* __restrict__ Qh, const _Float16* __restrict__ Kh,
    const _Float16* __restrict__ Vh, float* __restrict__ Xout) {
  constexpr int LDW = 72;  // halves; 144B rows, 16B aligned
  __shared__ __align__(16) _Float16 Kt[2][64 * LDW];    // Kt[key][dk]
  __shared__ __align__(16) _Float16 Vt[2][64 * LDW];    // Vt[dk][key]
  __shared__ __align__(16) _Float16 Pw[8][16 * LDW];    // per-wave P 16x64

  const int tid  = threadIdx.x;
  const int lane = tid & 31;
  const int wid  = tid >> 5;
  const int bb   = blockIdx.z;
  const int hh   = blockIdx.y;
  const int q0   = blockIdx.x * 128;
  const int fm   = lane & 15;
  const int hf   = lane >> 4;
  const int kb   = hf * 8;

  // load this wave's 16 Q rows as two A fragments (dk 0..31, 32..63),
  // pre-scaled by 1/sqrt(DK)=0.125 (exact in f16)
  const size_t qrow = (size_t)(bb * S_DIM + q0 + wid * 16 + fm);
  const _Float16* qp = Qh + qrow * D_DIM + hh * DK_DIM;
  v16h qa0 = frag16(qp + kb);
  v16h qa1 = frag16(qp + 32 + kb);
#pragma unroll
  for (int i = 0; i < 16; ++i) {
    qa0[i] *= (_Float16)0.125f;
    qa1[i] *= (_Float16)0.125f;
  }

  float rm[8], rl[8];
  v8f o[4];
#pragma unroll
  for (int v = 0; v < 8; ++v) { rm[v] = -1e30f; rl[v] = 0.0f; }
#pragma unroll
  for (int j = 0; j < 4; ++j)
#pragma unroll
    for (int v = 0; v < 8; ++v) o[j][v] = 0.0f;

  const int skey = tid >> 2;        // 0..63
  const int sc   = (tid & 3) * 16;  // dk chunk of 16

  auto stage = [&](int buf, int kt) {
    const size_t srow = (size_t)(bb * S_DIM + kt * 64 + skey) * D_DIM +
                        hh * DK_DIM + sc;
    // K tile: async global->LDS copy (2 x b128 per lane, ASYNCcnt path)
    async_g2l_b128(&Kt[buf][skey * LDW + sc],     Kh + srow);
    async_g2l_b128(&Kt[buf][skey * LDW + sc + 8], Kh + srow + 8);
    // V tile: load f16, transpose to Vt[dk][key] via b16 scatter
    v8h y0 = *(const v8h*)(Vh + srow);
    v8h y1 = *(const v8h*)(Vh + srow + 8);
#pragma unroll
    for (int d = 0; d < 8; ++d) {
      Vt[buf][(sc + d) * LDW + skey]     = y0[d];
      Vt[buf][(sc + 8 + d) * LDW + skey] = y1[d];
    }
  };

  stage(0, 0);

  int cur = 0;
  for (int kt = 0; kt < S_DIM / 64; ++kt) {
    wait_asynccnt0();   // own async K-copy into buf[cur] done
    __syncthreads();    // everyone's stores to buf[cur] visible
    if (kt + 1 < S_DIM / 64) stage(cur ^ 1, kt + 1);  // overlaps compute

    // ---- scores: S = Q(16x64) * K^T -> 4 tiles of 16 keys, 2 k-steps each
    v8f s[4];
#pragma unroll
    for (int j = 0; j < 4; ++j)
#pragma unroll
      for (int v = 0; v < 8; ++v) s[j][v] = 0.0f;
#pragma unroll
    for (int j = 0; j < 4; ++j) {
      const v16h bk0 = frag16(&Kt[cur][(j * 16 + fm) * LDW + kb]);
      const v16h bk1 = frag16(&Kt[cur][(j * 16 + fm) * LDW + 32 + kb]);
      s[j] = wmma_f16(qa0, bk0, s[j]);
      s[j] = wmma_f16(qa1, bk1, s[j]);
    }

    // ---- online softmax (per row = (hf, v); reduce across 16-lane half)
#pragma unroll
    for (int v = 0; v < 8; ++v) {
      float x = fmaxf(fmaxf(s[0][v], s[1][v]), fmaxf(s[2][v], s[3][v]));
      x = fmaxf(x, shfl_xor16<1>(x));
      x = fmaxf(x, shfl_xor16<2>(x));
      x = fmaxf(x, shfl_xor16<4>(x));
      x = fmaxf(x, shfl_xor16<8>(x));
      const float nm = fmaxf(rm[v], x);
      const float corr = __expf(rm[v] - nm);
      rm[v] = nm;
      float sum = 0.0f;
#pragma unroll
      for (int j = 0; j < 4; ++j) {
        const float p = __expf(s[j][v] - nm);
        s[j][v] = p;
        sum += p;
      }
      sum += shfl_xor16<1>(sum);
      sum += shfl_xor16<2>(sum);
      sum += shfl_xor16<4>(sum);
      sum += shfl_xor16<8>(sum);
      rl[v] = rl[v] * corr + sum;
#pragma unroll
      for (int j = 0; j < 4; ++j) o[j][v] *= corr;
    }

    // ---- C-layout P -> LDS -> A-layout fragments (per-wave buffer,
    //      same-wave DS ops are in-order; compiler fence keeps order)
#pragma unroll
    for (int j = 0; j < 4; ++j)
#pragma unroll
      for (int v = 0; v < 8; ++v)
        Pw[wid][(hf * 8 + v) * LDW + j * 16 + fm] = (_Float16)s[j][v];
    asm volatile("" ::: "memory");
    const v16h pa0 = frag16(&Pw[wid][fm * LDW + kb]);
    const v16h pa1 = frag16(&Pw[wid][fm * LDW + 32 + kb]);

    // ---- O += P(16x64) * V(64x64): 4 dk tiles x 2 k-steps
#pragma unroll
    for (int j = 0; j < 4; ++j) {
      const v16h bv0 = frag16(&Vt[cur][(j * 16 + fm) * LDW + kb]);
      const v16h bv1 = frag16(&Vt[cur][(j * 16 + fm) * LDW + 32 + kb]);
      o[j] = wmma_f16(pa0, bv0, o[j]);
      o[j] = wmma_f16(pa1, bv1, o[j]);
    }
    cur ^= 1;
  }

  // ---- finalize: divide by row sum, write [token, h*64+dk] f32
#pragma unroll
  for (int v = 0; v < 8; ++v) {
    const float inv = 1.0f / rl[v];
    const size_t row = (size_t)(bb * S_DIM + q0 + wid * 16 + hf * 8 + v);
#pragma unroll
    for (int j = 0; j < 4; ++j) {
      const int col = hh * DK_DIM + j * 16 + fm;
      Xout[row * D_DIM + col] = o[j][v] * inv;
    }
  }
}

// ---------------------------------------------------------------------------
extern "C" void kernel_launch(void* const* d_in, const int* in_sizes, int n_in,
                              void* d_out, int out_size, void* d_ws,
                              size_t ws_size, hipStream_t stream) {
  (void)in_sizes; (void)n_in; (void)out_size; (void)ws_size;
  const float* query = (const float*)d_in[0];
  const float* key_  = (const float*)d_in[1];
  const float* value = (const float*)d_in[2];
  const float* fc_w  = (const float*)d_in[3];
  const float* Wq    = (const float*)d_in[4];
  const float* bq    = (const float*)d_in[5];
  const float* Wk    = (const float*)d_in[6];
  const float* bk    = (const float*)d_in[7];
  const float* Wv    = (const float*)d_in[8];
  const float* bv    = (const float*)d_in[9];
  const float* Wo    = (const float*)d_in[10];
  const float* bo    = (const float*)d_in[11];
  float* out = (float*)d_out;

  char* ws = (char*)d_ws;
  const size_t half_mat = (size_t)NTOK * D_DIM * sizeof(_Float16);
  _Float16* Qh = (_Float16*)(ws);
  _Float16* Kh = (_Float16*)(ws + half_mat);
  _Float16* Vh = (_Float16*)(ws + 2 * half_mat);
  float*    Xa = (float*)   (ws + 3 * half_mat);

  dim3 gg(D_DIM / 128, NTOK / 128);
  gemm_wmma<1, 1><<<gg, 256, 0, stream>>>(query, Wq, fc_w, bq, (void*)Qh,
                                          NTOK, D_DIM, D_DIM);
  gemm_wmma<1, 0><<<gg, 256, 0, stream>>>(key_, Wk, nullptr, bk, (void*)Kh,
                                          NTOK, D_DIM, D_DIM);
  gemm_wmma<1, 0><<<gg, 256, 0, stream>>>(value, Wv, nullptr, bv, (void*)Vh,
                                          NTOK, D_DIM, D_DIM);
  attn_wmma<<<dim3(S_DIM / 128, H_DIM, 4), 256, 0, stream>>>(Qh, Kh, Vh, Xa);
  gemm_wmma<0, 0><<<gg, 256, 0, stream>>>(Xa, Wo, nullptr, bo, (void*)out,
                                          NTOK, D_DIM, D_DIM);
}
